// DetectionLoss_19997367730581
// MI455X (gfx1250) — compile-verified, matrix operand order checked by reading, not run on hardware
//
#include <hip/hip_runtime.h>
#include <hip/hip_bf16.h>

typedef __attribute__((ext_vector_type(16))) _Float16 v16h;
typedef __attribute__((ext_vector_type(8)))  float    v8f;

#define BDIM 32
#define NDIM 32768
#define CDIM 81
#define MDIM 40
#define APB  128     // anchors per block
#define THREADS 128
#define NEG_FALLBACK 1638   // int(N * 0.05)

__device__ __forceinline__ unsigned f2sort(float f) {
    unsigned u = __float_as_uint(f);
    return (u & 0x80000000u) ? ~u : (u | 0x80000000u);
}
__device__ __forceinline__ float sort2f(unsigned u) {
    unsigned b = (u & 0x80000000u) ? (u & 0x7FFFFFFFu) : ~u;
    return __uint_as_float(b);
}

__global__ void ssd_init_kernel(float* facc, int* iacc, float* tot, int* tp) {
    int i = threadIdx.x;
    if (i < 2 * BDIM) { facc[i] = 0.0f; iacc[i] = 0; }
    if (i < 2) tot[i] = 0.0f;
    if (i == 0) *tp = 0;
}

// Fused: IoU matching + SmoothL1 + CE (logsumexp via WMMA) per 128-anchor tile.
__global__ __launch_bounds__(THREADS) void ssd_match_ce_kernel(
    const float* __restrict__ logits,   // [B,N,C]
    const float* __restrict__ pred,     // [B,N,4] cxcywh
    const float* __restrict__ gtb,      // [B,M,4] xyxy
    const int*   __restrict__ gtl,      // [B,M]
    const float* __restrict__ defs,     // [N,4] xyxy
    float* __restrict__ negce,          // [B,N] neg CE or -1
    float* __restrict__ facc,           // [B*2] loc_sum, pos_ce_sum
    int*   __restrict__ iacc)           // [B*2] num_pos, num_neg
{
    __shared__ __align__(16) float tile[APB * CDIM];   // 41472 B
    __shared__ float sgtb[MDIM * 4];
    __shared__ int   sgtl[MDIM];
    __shared__ float smax[APB];
    __shared__ float ssum[APB];
    __shared__ float s_loc, s_pce;
    __shared__ int   s_np, s_nn;

    const int img   = blockIdx.y;
    const int abase = blockIdx.x * APB;
    const int tid   = threadIdx.x;
    const int wave  = tid >> 5;
    const int lane  = tid & 31;

    if (tid == 0) { s_loc = 0.0f; s_pce = 0.0f; s_np = 0; s_nn = 0; }
    for (int i = tid; i < MDIM * 4; i += THREADS) sgtb[i] = gtb[img * MDIM * 4 + i];
    for (int i = tid; i < MDIM;     i += THREADS) sgtl[i] = gtl[img * MDIM + i];

    // 128 consecutive anchors => contiguous 128*81 floats; coalesced b128 loads.
    {
        const float4* src4 = (const float4*)(logits + ((size_t)img * NDIM + abase) * CDIM);
        float4* t4 = (float4*)tile;
        for (int i = tid; i < (APB * CDIM) / 4; i += THREADS) t4[i] = src4[i];  // 10368 % 4 == 0
    }
    __syncthreads();

    const int a_local = tid;            // this thread's anchor within the tile
    const int a = abase + a_local;

    // ---- IoU matching (first-max tie-break == argmax) ----
    float4 d = ((const float4*)defs)[a];
    float area_d = (d.z - d.x) * (d.w - d.y);
    float best = -1.0f; int bm = 0;
    for (int m = 0; m < MDIM; ++m) {
        float g0 = sgtb[m*4+0], g1 = sgtb[m*4+1], g2 = sgtb[m*4+2], g3 = sgtb[m*4+3];
        float lx = fmaxf(g0, d.x), ly = fmaxf(g1, d.y);
        float rx = fminf(g2, d.z), ry = fminf(g3, d.w);
        float w  = fmaxf(rx - lx, 0.0f), h = fmaxf(ry - ly, 0.0f);
        float inter = w * h;
        float ag = (g2 - g0) * (g3 - g1);
        float iou = inter / (ag + area_d - inter + 1e-10f);
        if (iou > best) { best = iou; bm = m; }
    }
    bool pos = best >= 0.5f;
    bool neg = (best < 0.4f) && !pos;
    int  target = pos ? (sgtl[bm] + 1) : 0;

    // ---- SmoothL1 localization (positives only) ----
    float loc = 0.0f;
    {
        float4 p = ((const float4*)pred)[(size_t)img * NDIM + a];
        float q[4] = { p.x - 0.5f * p.z, p.y - 0.5f * p.w,
                       p.x + 0.5f * p.z, p.y + 0.5f * p.w };
        if (pos) {
            #pragma unroll
            for (int c = 0; c < 4; ++c) {
                float dd = fabsf(q[c] - sgtb[bm * 4 + c]);
                loc += (dd < 1.0f) ? 0.5f * dd * dd : (dd - 0.5f);
            }
        }
    }

    // ---- per-anchor class max (stride 81 => gcd(81,64)=1, conflict-free) ----
    {
        const float* row = tile + a_local * CDIM;
        float mx = row[0];
        for (int c = 1; c < CDIM; ++c) mx = fmaxf(mx, row[c]);
        smax[a_local] = mx;
    }
    __syncthreads();

    // ---- sum(exp) via V_WMMA_F32_16X16X32_F16; 2 subtiles of 16 anchors/wave ----
    const bool hi   = (lane >= 16);
    const int  mrow = lane & 15;
    v16h bmat;
    #pragma unroll
    for (int i = 0; i < 16; ++i) bmat[i] = (_Float16)1.0f;   // all-ones B (layout-invariant)

    #pragma unroll
    for (int s = 0; s < 2; ++s) {
        const int rowi = wave * 32 + s * 16 + mrow;
        const float* rowp = tile + rowi * CDIM;
        const float mx = smax[rowi];
        v8f c = {};
        #pragma unroll
        for (int t = 0; t < 3; ++t) {            // classes 0..95, pad>=81 -> 0
            const int kb0 = t * 32 + (hi ? 8 : 0);   // ISA 16-bit A 16x32 layout
            const int kb1 = kb0 + 16;
            v16h amat;
            #pragma unroll
            for (int i = 0; i < 8; ++i) {
                int k0 = kb0 + i, k1 = kb1 + i;
                amat[i]     = (k0 < CDIM) ? (_Float16)__expf(rowp[k0] - mx) : (_Float16)0.0f;
                amat[8 + i] = (k1 < CDIM) ? (_Float16)__expf(rowp[k1] - mx) : (_Float16)0.0f;
            }
            c = __builtin_amdgcn_wmma_f32_16x16x32_f16(
                    false, amat, false, bmat, (short)0, c, false, false);
        }
        // Column 0 of D: lane 0 holds M=0..7 (VGPR 0..7), lane 16 holds M=8..15.
        if ((lane & 15) == 0) {
            int rbase = wave * 32 + s * 16 + (hi ? 8 : 0);
            #pragma unroll
            for (int r = 0; r < 8; ++r) ssum[rbase + r] = c[r];
        }
    }
    __syncthreads();

    float lse = smax[a_local] + __logf(ssum[a_local]);
    float ce  = lse - tile[a_local * CDIM + target];

    negce[(size_t)img * NDIM + a] = neg ? ce : -1.0f;
    if (pos) { atomicAdd(&s_loc, loc); atomicAdd(&s_pce, ce); atomicAdd(&s_np, 1); }
    if (neg) { atomicAdd(&s_nn, 1); }
    __syncthreads();
    if (tid == 0) {
        if (s_loc != 0.0f) atomicAdd(&facc[img * 2 + 0], s_loc);
        if (s_pce != 0.0f) atomicAdd(&facc[img * 2 + 1], s_pce);
        if (s_np) atomicAdd(&iacc[img * 2 + 0], s_np);
        if (s_nn) atomicAdd(&iacc[img * 2 + 1], s_nn);
    }
}

// Per-image exact top-k sum via 4-pass (8 bits each) radix select on sortable bits.
__global__ __launch_bounds__(256) void ssd_topk_kernel(
    const float* __restrict__ negce,
    const float* __restrict__ facc,
    const int*   __restrict__ iacc,
    float* __restrict__ tot,   // [0]=loc_total [1]=conf_total
    int*   __restrict__ tp)
{
    __shared__ int hist[256];
    __shared__ unsigned s_pref;
    __shared__ int s_kk;
    __shared__ float red[256];

    const int img = blockIdx.x;
    const int tid = threadIdx.x;
    const float* vals = negce + (size_t)img * NDIM;

    const int np = iacc[img * 2 + 0];
    const int nn = iacc[img * 2 + 1];
    const int k  = (np > 0) ? min(np * 3, nn) : min(NEG_FALLBACK, nn);

    float neg_sum = 0.0f;
    if (k > 0) {                                  // uniform across block
        if (tid == 0) { s_pref = 0u; s_kk = k; }
        __syncthreads();
        for (int p = 0; p < 4; ++p) {
            const int shift = 24 - 8 * p;
            hist[tid] = 0;
            __syncthreads();
            const unsigned pref  = s_pref;
            const unsigned hmask = (p == 0) ? 0u : (0xFFFFFFFFu << (shift + 8));
            for (int i = tid; i < NDIM; i += 256) {
                unsigned u = f2sort(vals[i]);
                if ((u & hmask) == (pref & hmask))
                    atomicAdd(&hist[(u >> shift) & 255], 1);
            }
            __syncthreads();
            if (tid == 0) {
                int kk = s_kk;
                int b;
                for (b = 255; b > 0; --b) {
                    if (kk <= hist[b]) break;
                    kk -= hist[b];
                }
                s_kk = kk;
                s_pref = pref | ((unsigned)b << shift);
            }
            __syncthreads();
        }
        const unsigned kth = s_pref;   // exact sortable bits of k-th largest
        const int remaining = s_kk;    // ties taken at the threshold
        float partial = 0.0f;
        for (int i = tid; i < NDIM; i += 256) {
            float v = vals[i];
            if (f2sort(v) > kth) partial += v;
        }
        red[tid] = partial;
        __syncthreads();
        for (int off = 128; off > 0; off >>= 1) {
            if (tid < off) red[tid] += red[tid + off];
            __syncthreads();
        }
        if (tid == 0) neg_sum = red[0] + (float)remaining * sort2f(kth);
    }
    if (tid == 0) {
        atomicAdd(&tot[0], facc[img * 2 + 0]);
        atomicAdd(&tot[1], facc[img * 2 + 1] + neg_sum);
        atomicAdd(tp, np);
    }
}

__global__ void ssd_final_kernel(const float* __restrict__ tot,
                                 const int* __restrict__ tp,
                                 float* __restrict__ out)
{
    float loc_total = tot[0], conf_total = tot[1];
    int t = *tp;
    float denom = (float)((t > 1) ? t : 1);
    float loc_norm  = (t > 0) ? loc_total / denom : 0.0f;
    float conf_norm = (t > 0) ? conf_total / denom
                              : ((conf_total > 0.0f)
                                     ? conf_total / (float)(BDIM * NDIM) : 0.0f);
    out[0] = loc_norm + conf_norm;
    out[1] = loc_norm;
    out[2] = conf_norm;
}

extern "C" void kernel_launch(void* const* d_in, const int* in_sizes, int n_in,
                              void* d_out, int out_size, void* d_ws, size_t ws_size,
                              hipStream_t stream) {
    (void)in_sizes; (void)n_in; (void)out_size; (void)ws_size;
    const float* logits = (const float*)d_in[0];  // [B,N,C]
    const float* pred   = (const float*)d_in[1];  // [B,N,4]
    const float* gtb    = (const float*)d_in[2];  // [B,M,4]
    const int*   gtl    = (const int*)d_in[3];    // [B,M]
    const float* defs   = (const float*)d_in[4];  // [N,4]
    float* out = (float*)d_out;

    float* negce = (float*)d_ws;                        // B*N floats = 4 MB
    float* facc  = negce + (size_t)BDIM * NDIM;         // 2*B floats
    int*   iacc  = (int*)(facc + 2 * BDIM);             // 2*B ints
    float* tot   = (float*)(iacc + 2 * BDIM);           // 2 floats
    int*   tp    = (int*)(tot + 2);                     // 1 int

    ssd_init_kernel<<<1, 64, 0, stream>>>(facc, iacc, tot, tp);
    dim3 grid(NDIM / APB, BDIM);
    ssd_match_ce_kernel<<<grid, THREADS, 0, stream>>>(
        logits, pred, gtb, gtl, defs, negce, facc, iacc);
    ssd_topk_kernel<<<BDIM, 256, 0, stream>>>(negce, facc, iacc, tot, tp);
    ssd_final_kernel<<<1, 1, 0, stream>>>(tot, tp, out);
}
